// LSTMBaseline_85109071937819
// MI455X (gfx1250) — compile-verified, matrix operand order checked by reading, not run on hardware
//
#include <hip/hip_runtime.h>
#include <hip/hip_bf16.h>
#include <math.h>

// ---------------------------------------------------------------------------
// Model dims (reference): V,D,H,B,S = 50257,768,2048,8,512
// ---------------------------------------------------------------------------
#define C_B   8
#define C_S   512
#define C_D   768
#define C_H   2048
#define C_V   50257
#define C_4H  8192

typedef __bf16 bf16_t;
typedef __attribute__((ext_vector_type(16))) __bf16 v16bf;
typedef __attribute__((ext_vector_type(8)))  float  v8f;
typedef __attribute__((ext_vector_type(4)))  int    v4i;

union FragB16 { v16bf v; uint4 q[2]; };

// ---------------------------------------------------------------------------
// CDNA5 async global->LDS copy (ASYNCcnt path), with sync fallback
// ---------------------------------------------------------------------------
#if defined(__has_builtin)
#  if __has_builtin(__builtin_amdgcn_global_load_async_to_lds_b128)
#    define HAVE_ASYNC_LDS 1
#  endif
#endif
#ifndef HAVE_ASYNC_LDS
#  define HAVE_ASYNC_LDS 0
#endif

typedef __attribute__((address_space(1))) v4i gv4i_t;   // global v4i
typedef __attribute__((address_space(3))) v4i lv4i_t;   // LDS v4i

__device__ __forceinline__ void async_cp16(const bf16_t* g, bf16_t* l) {
#if HAVE_ASYNC_LDS
  // -> global_load_async_to_lds_b128 (16 bytes per lane)
  __builtin_amdgcn_global_load_async_to_lds_b128((gv4i_t*)g, (lv4i_t*)l, 0, 0);
#else
  *(uint4*)l = *(const uint4*)g;
#endif
}

__device__ __forceinline__ void wait_async_lds() {
#if HAVE_ASYNC_LDS
#  if __has_builtin(__builtin_amdgcn_s_wait_asynccnt)
  __builtin_amdgcn_s_wait_asynccnt(0);
#  endif
#endif
}

// ---------------------------------------------------------------------------
// Elementwise helpers
// ---------------------------------------------------------------------------
__global__ void cvt_f32_to_bf16(const float* __restrict__ in, bf16_t* __restrict__ out,
                                size_t n) {
  size_t i = (size_t)blockIdx.x * blockDim.x + threadIdx.x;
  if (i < n) out[i] = (bf16_t)in[i];
}

__global__ void gather_embed(const int* __restrict__ ids, const float* __restrict__ emb,
                             bf16_t* __restrict__ xbf) {
  size_t i = (size_t)blockIdx.x * blockDim.x + threadIdx.x;
  const size_t n = (size_t)C_B * C_S * C_D;
  if (i >= n) return;
  size_t row = i / C_D, d = i % C_D;
  int tok = ids[row];
  xbf[i] = (bf16_t)emb[(size_t)tok * C_D + d];
}

__global__ void add_bias_pair(const float* __restrict__ a, const float* __restrict__ b,
                              float* __restrict__ out, int n) {
  int i = blockIdx.x * blockDim.x + threadIdx.x;
  if (i < n) out[i] = a[i] + b[i];
}

__global__ void zero_state(bf16_t* __restrict__ hbf, float* __restrict__ c) {
  int i = blockIdx.x * blockDim.x + threadIdx.x;
  if (i < 16 * C_H) hbf[i] = (bf16_t)0.0f;     // 16 rows: 8 real + 8 zero-pad for WMMA
  if (i < C_B * C_H) c[i] = 0.0f;
}

// ---------------------------------------------------------------------------
// Generic bf16 GEMM:  C[M,N] = A[M,K] @ W[N,K]^T (+ bias[N])
// block = 128 threads (4 waves), block tile 128x64, K-step 32, double-buffered
// LDS staged via async global->LDS copies (ASYNCcnt).
// Wave w computes rows [w*32, w*32+32) as 2 M-subtiles x 4 N-subtiles
// (8 WMMA / K-step, B fragments reused across both M rows).
// Fragment layout per CDNA5 ISA 7.12.2 (16-bit A 16x32):
//   lanes 0-15  : M=lane,    K = {0..7} U {16..23}
//   lanes 16-31 : M=lane-16, K = {8..15} U {24..31}
// B (32x16, K x N): lane = column N, same K split.
// C/D (16x16 f32): lanes 0-15 N=lane M=v; lanes 16-31 N=lane-16 M=v+8.
// ---------------------------------------------------------------------------
#define BM 128
#define BN 64
#define BK 32

__launch_bounds__(128)
__global__ void gemm_bf16_wmma(const bf16_t* __restrict__ A, const bf16_t* __restrict__ W,
                               const float* __restrict__ bias,
                               float* __restrict__ Cf, bf16_t* __restrict__ Cbf,
                               int M, int N, int K) {
  __shared__ bf16_t lA[2][BM * BK];   // 2 x 8 KB
  __shared__ bf16_t lW[2][BN * BK];   // 2 x 4 KB

  const int tid  = threadIdx.x;
  const int wave = tid >> 5;
  const int lane = tid & 31;
  const int m0 = blockIdx.y * BM;
  const int n0 = blockIdx.x * BN;

  const int fr  = lane & 15;          // row (A) / col (B) within 16-tile
  const int lhi = lane >> 4;          // lane-half select
  const int kb  = lhi * 8;            // K base for this lane-half

  v8f acc[2][4] = {};

  // Cooperative tile loads:
  //  A tile 128x32: thread t copies full row t      (4 x 16B async)
  //  W tile  64x32: thread t copies half row t>>1   (2 x 16B async)
  const int ra   = tid;
  const int rw   = tid >> 1;
  const int wseg = (tid & 1) * 16;    // element offset within W row
  int am = m0 + ra; if (am >= M) am = M - 1;
  int wn = n0 + rw; if (wn >= N) wn = N - 1;
  const bf16_t* aptr = A + (size_t)am * K;
  const bf16_t* wptr = W + (size_t)wn * K;

  auto issue_tile = [&](int buf, int k0) {
    bf16_t* la = &lA[buf][ra * BK];
    bf16_t* lw = &lW[buf][rw * BK + wseg];
#pragma unroll
    for (int s = 0; s < 4; ++s) async_cp16(aptr + k0 + s * 8, la + s * 8);
#pragma unroll
    for (int s = 0; s < 2; ++s) async_cp16(wptr + k0 + wseg + s * 8, lw + s * 8);
  };

  issue_tile(0, 0);

  for (int k0 = 0; k0 < K; k0 += BK) {
    const int cur = (k0 / BK) & 1;
    wait_async_lds();                 // all async writes into lA/lW[cur] landed
    __syncthreads();                  // ... and visible to every wave
    if (k0 + BK < K) issue_tile(cur ^ 1, k0 + BK);   // overlap next tile copy

    FragB16 a0, a1;
    const bf16_t* ar0 = &lA[cur][(wave * 32 + fr) * BK];
    const bf16_t* ar1 = ar0 + 16 * BK;
    a0.q[0] = *(const uint4*)(ar0 + kb);
    a0.q[1] = *(const uint4*)(ar0 + kb + 16);
    a1.q[0] = *(const uint4*)(ar1 + kb);
    a1.q[1] = *(const uint4*)(ar1 + kb + 16);

#pragma unroll
    for (int nt = 0; nt < 4; ++nt) {
      FragB16 b;
      const bf16_t* br = &lW[cur][(nt * 16 + fr) * BK];
      b.q[0] = *(const uint4*)(br + kb);
      b.q[1] = *(const uint4*)(br + kb + 16);
      acc[0][nt] = __builtin_amdgcn_wmma_f32_16x16x32_bf16(
          false, a0.v, false, b.v, (short)0, acc[0][nt], false, false);
      acc[1][nt] = __builtin_amdgcn_wmma_f32_16x16x32_bf16(
          false, a1.v, false, b.v, (short)0, acc[1][nt], false, false);
    }
    __syncthreads();                  // done reading buf cur before reuse
  }

#pragma unroll
  for (int mi = 0; mi < 2; ++mi) {
#pragma unroll
    for (int nt = 0; nt < 4; ++nt) {
      const int n = n0 + nt * 16 + fr;
      if (n >= N) continue;
      const float bv = bias ? bias[n] : 0.0f;
#pragma unroll
      for (int v = 0; v < 8; ++v) {
        const int m = m0 + wave * 32 + mi * 16 + lhi * 8 + v;
        if (m >= M) continue;
        const float val = acc[mi][nt][v] + bv;
        const size_t o = (size_t)m * N + n;
        if (Cf)  Cf[o]  = val;
        if (Cbf) Cbf[o] = (bf16_t)val;
      }
    }
  }
}

// ---------------------------------------------------------------------------
// Recurrent step GEMM: gbuf[16,4H] = Hb[16,H] @ Whh[4H,H]^T
// One wave per 16-wide N tile; M=16 (rows 8..15 are zero padding).
// Whh (33.5 MB bf16) stays resident in the 192 MB global L2 across steps.
// ---------------------------------------------------------------------------
__launch_bounds__(32)
__global__ void lstm_step_gemm(const bf16_t* __restrict__ Hb,
                               const bf16_t* __restrict__ Whh,
                               float* __restrict__ gbuf) {
  const int lane = threadIdx.x;
  const int n0 = blockIdx.x * 16;
  const int fr  = lane & 15;
  const int lhi = lane >> 4;
  const int kb  = lhi * 8;

  const bf16_t* arow = Hb  + (size_t)fr * C_H;
  const bf16_t* brow = Whh + (size_t)(n0 + fr) * C_H;

  v8f acc = {};
  for (int k0 = 0; k0 < C_H; k0 += 32) {
    FragB16 a, b;
    a.q[0] = *(const uint4*)(arow + k0 + kb);
    a.q[1] = *(const uint4*)(arow + k0 + kb + 16);
    b.q[0] = *(const uint4*)(brow + k0 + kb);
    b.q[1] = *(const uint4*)(brow + k0 + kb + 16);
    acc = __builtin_amdgcn_wmma_f32_16x16x32_bf16(
        false, a.v, false, b.v, (short)0, acc, false, false);
  }
#pragma unroll
  for (int v = 0; v < 8; ++v) {
    const int m = lhi * 8 + v;
    gbuf[(size_t)m * C_4H + n0 + fr] = acc[v];
  }
}

// ---------------------------------------------------------------------------
// LSTM gate nonlinearity + state update (f32 VALU), PyTorch gate order i,f,g,o.
// ---------------------------------------------------------------------------
__global__ void lstm_gate(const float* __restrict__ gbuf, const float* __restrict__ xg,
                          float* __restrict__ c, bf16_t* __restrict__ hbf,
                          bf16_t* __restrict__ hsbf, int t) {
  const int i = blockIdx.x * blockDim.x + threadIdx.x;
  if (i >= C_B * C_H) return;
  const int b = i >> 11;            // H = 2048
  const int j = i & (C_H - 1);
  const size_t xo = ((size_t)b * C_S + t) * C_4H;
  const size_t go = (size_t)b * C_4H;

  const float gi = gbuf[go + j          ] + xg[xo + j          ];
  const float gf = gbuf[go + j + C_H    ] + xg[xo + j + C_H    ];
  const float gc = gbuf[go + j + 2 * C_H] + xg[xo + j + 2 * C_H];
  const float gq = gbuf[go + j + 3 * C_H] + xg[xo + j + 3 * C_H];

  const float ig = 1.0f / (1.0f + __expf(-gi));
  const float fg = 1.0f / (1.0f + __expf(-gf));
  const float cg = tanhf(gc);
  const float og = 1.0f / (1.0f + __expf(-gq));

  const float cn = fg * c[i] + ig * cg;
  c[i] = cn;
  const float hn = og * tanhf(cn);

  hbf[(size_t)b * C_H + j] = (bf16_t)hn;                    // next-step A matrix
  hsbf[((size_t)b * C_S + t) * C_H + j] = (bf16_t)hn;       // [B,S,H] history
}

// ---------------------------------------------------------------------------
// Host-side orchestration
// ---------------------------------------------------------------------------
extern "C" void kernel_launch(void* const* d_in, const int* in_sizes, int n_in,
                              void* d_out, int out_size, void* d_ws, size_t ws_size,
                              hipStream_t stream) {
  const int*   ids    = (const int*)  d_in[0];
  const float* emb    = (const float*)d_in[1];
  const float* w_ih   = (const float*)d_in[2];
  const float* w_hh   = (const float*)d_in[3];
  const float* b_ih   = (const float*)d_in[4];
  const float* b_hh   = (const float*)d_in[5];
  const float* proj_w = (const float*)d_in[6];
  const float* proj_b = (const float*)d_in[7];
  const float* head_w = (const float*)d_in[8];
  float* out = (float*)d_out;
  (void)in_sizes; (void)n_in; (void)out_size; (void)ws_size;

  char* ws = (char*)d_ws;
  size_t off = 0;
  auto alloc = [&](size_t bytes) -> void* {
    void* p = ws + off;
    off = (off + bytes + 255) & ~(size_t)255;
    return p;
  };

  bf16_t* xbf    = (bf16_t*)alloc((size_t)C_B * C_S * C_D * sizeof(bf16_t));
  bf16_t* wihb   = (bf16_t*)alloc((size_t)C_4H * C_D * sizeof(bf16_t));
  bf16_t* whhb   = (bf16_t*)alloc((size_t)C_4H * C_H * sizeof(bf16_t));
  bf16_t* projb_ = (bf16_t*)alloc((size_t)C_D * C_H * sizeof(bf16_t));
  bf16_t* headb  = (bf16_t*)alloc((size_t)C_V * C_D * sizeof(bf16_t));
  float*  xg     = (float*) alloc((size_t)C_B * C_S * C_4H * sizeof(float));
  bf16_t* hsbf   = (bf16_t*)alloc((size_t)C_B * C_S * C_H * sizeof(bf16_t));
  bf16_t* ybf    = (bf16_t*)alloc((size_t)C_B * C_S * C_D * sizeof(bf16_t));
  float*  gbuf   = (float*) alloc((size_t)16 * C_4H * sizeof(float));
  bf16_t* hbf    = (bf16_t*)alloc((size_t)16 * C_H * sizeof(bf16_t));
  float*  cst    = (float*) alloc((size_t)C_B * C_H * sizeof(float));
  float*  bsum   = (float*) alloc((size_t)C_4H * sizeof(float));

  auto cvt = [&](const float* src, bf16_t* dst, size_t n) {
    cvt_f32_to_bf16<<<(unsigned)((n + 255) / 256), 256, 0, stream>>>(src, dst, n);
  };

  // 1) Precision cast of all GEMM operands to bf16
  cvt(w_ih,   wihb,   (size_t)C_4H * C_D);
  cvt(w_hh,   whhb,   (size_t)C_4H * C_H);
  cvt(proj_w, projb_, (size_t)C_D * C_H);
  cvt(head_w, headb,  (size_t)C_V * C_D);

  // 2) Embedding gather (f32 table -> bf16 activations)
  {
    size_t n = (size_t)C_B * C_S * C_D;
    gather_embed<<<(unsigned)((n + 255) / 256), 256, 0, stream>>>(ids, emb, xbf);
  }

  // 3) bsum = b_ih + b_hh
  add_bias_pair<<<(C_4H + 255) / 256, 256, 0, stream>>>(b_ih, b_hh, bsum, C_4H);

  // 4) xg = x @ w_ih^T + bsum   [4096, 8192]
  {
    dim3 g((C_4H + BN - 1) / BN, (C_B * C_S + BM - 1) / BM);
    gemm_bf16_wmma<<<g, 128, 0, stream>>>(xbf, wihb, bsum, xg, nullptr,
                                          C_B * C_S, C_4H, C_D);
  }

  // 5) Recurrence (sequential in S; per-step GEMM on the WMMA pipes)
  zero_state<<<(16 * C_H + 255) / 256, 256, 0, stream>>>(hbf, cst);
  for (int t = 0; t < C_S; ++t) {
    lstm_step_gemm<<<C_4H / 16, 32, 0, stream>>>(hbf, whhb, gbuf);
    lstm_gate<<<(C_B * C_H + 255) / 256, 256, 0, stream>>>(gbuf, xg, cst, hbf, hsbf, t);
  }

  // 6) y = hs @ proj_w^T + proj_b  -> bf16 for head GEMM
  {
    dim3 g((C_D + BN - 1) / BN, (C_B * C_S + BM - 1) / BM);
    gemm_bf16_wmma<<<g, 128, 0, stream>>>(hsbf, projb_, proj_b, nullptr, ybf,
                                          C_B * C_S, C_D, C_H);
  }

  // 7) logits = y @ head_w^T   [4096, 50257] f32 output
  {
    dim3 g((C_V + BN - 1) / BN, (C_B * C_S + BM - 1) / BM);
    gemm_bf16_wmma<<<g, 128, 0, stream>>>(ybf, headb, nullptr, out, nullptr,
                                          C_B * C_S, C_V, C_D);
  }
}
